// BridgeLayer3_xb_72000831750147
// MI455X (gfx1250) — compile-verified, matrix operand order checked by reading, not run on hardware
//
#include <hip/hip_runtime.h>
#include <math.h>

// ---------------------------------------------------------------------------
// CDNA5 / gfx1250 implementation of the VMamba "BridgeLayer3".
// GEMMs: one 16x16 tile per wave via v_wmma_f32_16x16x32_f16, operand tiles
// staged into LDS by the Tensor Data Mover (tensor_load_to_lds + D#
// descriptors built per CDNA5 ISA ch.8), fragments read back via ds_load.
// Scan: 16 lanes per channel (one per SSM state), wave32 __shfl_xor reduce.
// ---------------------------------------------------------------------------

typedef __attribute__((ext_vector_type(16))) _Float16 v16h;
typedef __attribute__((ext_vector_type(8)))  float    v8f;
typedef __attribute__((ext_vector_type(4)))  unsigned int v4u;
typedef __attribute__((ext_vector_type(8)))  int      v8i;
typedef __attribute__((ext_vector_type(4)))  int      v4i;

#define NST 16  // N_STATE

__device__ __forceinline__ int imin(int a, int b) { return a < b ? a : b; }
__device__ __forceinline__ int imax(int a, int b) { return a > b ? a : b; }

// ---------------------------------------------------------------------------
// TDM 2-D tile load: global (f32 elements, contiguous dim0, strided dim1)
// -> LDS, laid out as tile1 rows x tile0 contiguous floats.
// Descriptor bitfields per CDNA5 ISA ch.8 (D# group0 / group1).
// Hardware zero-fills reads beyond (rem0, rem1), giving free edge padding.
// ---------------------------------------------------------------------------
__device__ __forceinline__ void tdm_load_2d(const float* gbase, long long elem_off,
                                            unsigned lds_byte_off,
                                            int rem0, int rem1,
                                            int tile0, int tile1,
                                            long long stride1 /*elements*/)
{
    unsigned long long ga = (unsigned long long)(const void*)gbase +
                            (unsigned long long)elem_off * 4ull;
    unsigned d0 = (unsigned)imax(rem0, 0);
    unsigned d1 = (unsigned)imax(rem1, 0);
    unsigned long long sb = (unsigned long long)stride1;   // in data_size units

    v4u g0;
    g0[0] = 1u;                                    // count=1, user mode
    g0[1] = lds_byte_off;                          // LDS byte address
    g0[2] = (unsigned)(ga & 0xffffffffull);        // global_addr[31:0]
    g0[3] = (unsigned)((ga >> 32) & 0x01ffffffull) // global_addr[56:32]
            | (2u << 30);                          // type = 2 ("image")

    v8i g1;
    g1[0] = (int)(2u << 16);                       // data_size=2 -> 4 bytes
    g1[1] = (int)((d0 & 0xffffu) << 16);           // tensor_dim0[15:0]
    g1[2] = (int)((d0 >> 16) | ((d1 & 0xffffu) << 16));   // dim0[31:16]|dim1[15:0]
    g1[3] = (int)((d1 >> 16) | ((unsigned)tile0 << 16));  // dim1[31:16]|tile_dim0
    g1[4] = (int)((unsigned)tile1);                // tile_dim1 (tile_dim2 = 0)
    g1[5] = (int)(unsigned)(sb & 0xffffffffull);   // tensor_dim0_stride[31:0]
    g1[6] = (int)(unsigned)((sb >> 32) & 0xffffull); // stride[47:32]
    g1[7] = 0;

    v4i z4 = {};
    v8i z8 = {};
    __builtin_amdgcn_tensor_load_to_lds(g0, g1, z4, z4, z8, 0);
}

// ---------------------------------------------------------------------------
// TDM-staged strided GEMM:  O(m,n) = sum_k A(m,k)*B(n,k)  [+ Res(m,n)]
// A: if aRM (k contiguous): A[m*Abig + k]  else (m contiguous): A[k*Abig + m]
// B: if bRM (k contiguous): B[n*Bbig + k]  else (n contiguous): B[k*Bbig + n]
// One wave per block; 16x16 output tile; K swept in steps of 32.
// ---------------------------------------------------------------------------
__global__ void wmma_gemm_tdm(const float* __restrict__ A,
                              const float* __restrict__ B,
                              float* __restrict__ O,
                              const float* __restrict__ Res,
                              int M, int N, int K,
                              long long Abig, int aRM,
                              long long Bbig, int bRM,
                              long long Os_m, long long Os_n)
{
    __shared__ float ldsA[16 * 32];
    __shared__ float ldsB[16 * 32];

    const int lane  = threadIdx.x & 31;
    const int half  = lane >> 4;
    const int lm    = lane & 15;
    const int kbase = half << 3;
    const int tm = blockIdx.x * 16;
    const int tn = blockIdx.y * 16;

    // Low 32 bits of a generic LDS pointer are the LDS byte offset
    // (see ISA 10.2).
    const unsigned offA = (unsigned)(unsigned long long)(const void*)&ldsA[0];
    const unsigned offB = (unsigned)(unsigned long long)(const void*)&ldsB[0];

    v8f acc = {};
    for (int k0 = 0; k0 < K; k0 += 32) {
        if (k0 > 0)
            asm volatile("s_wait_dscnt 0" ::: "memory");  // LDS reads done before overwrite

        if (aRM)   // tile: dim0 = k (32, contiguous), dim1 = m (16, stride Abig)
            tdm_load_2d(A, (long long)tm * Abig + k0, offA,
                        K - k0, M - tm, 32, 16, Abig);
        else       // tile: dim0 = m (16, contiguous), dim1 = k (32, stride Abig)
            tdm_load_2d(A, (long long)k0 * Abig + tm, offA,
                        M - tm, K - k0, 16, 32, Abig);

        if (bRM)
            tdm_load_2d(B, (long long)tn * Bbig + k0, offB,
                        K - k0, N - tn, 32, 16, Bbig);
        else
            tdm_load_2d(B, (long long)k0 * Bbig + tn, offB,
                        N - tn, K - k0, 16, 32, Bbig);

        __builtin_amdgcn_s_wait_tensorcnt(0);

        v16h a, b;
#pragma unroll
        for (int j = 0; j < 8; ++j) {
            const int kj = (j < 4) ? (kbase + 2 * j) : (16 + kbase + 2 * (j - 4));
#pragma unroll
            for (int t = 0; t < 2; ++t) {
                const int kk = kj + t;
                float av = aRM ? ldsA[lm * 32 + kk] : ldsA[kk * 16 + lm];
                float bv = bRM ? ldsB[lm * 32 + kk] : ldsB[kk * 16 + lm];
                a[2 * j + t] = (_Float16)av;
                b[2 * j + t] = (_Float16)bv;
            }
        }
        acc = __builtin_amdgcn_wmma_f32_16x16x32_f16(
                  false, a, false, b, (short)0, acc, false, false);
    }

    const int n = tn + lm;
    if (n < N) {
#pragma unroll
        for (int r = 0; r < 8; ++r) {
            const int m = tm + (half << 3) + r;
            if (m < M) {
                long long oi = (long long)m * Os_m + (long long)n * Os_n;
                float v = acc[r];
                if (Res) v += Res[oi];
                O[oi] = v;
            }
        }
    }
}

// ---------------------------------------------------------------------------
// Fallback generic strided GEMM (no TDM): clamped unconditional loads with
// value masking -> no exec divergence, no per-load waits. Kept compiled.
// ---------------------------------------------------------------------------
__global__ void wmma_gemm_f16(const float* __restrict__ A,
                              const float* __restrict__ B,
                              float* __restrict__ O,
                              const float* __restrict__ Res,
                              int M, int N, int K,
                              long long As_m, long long As_k,
                              long long Bs_n, long long Bs_k,
                              long long Os_m, long long Os_n)
{
    const int lane  = threadIdx.x & 31;
    const int half  = lane >> 4;
    const int lm    = lane & 15;
    const int kbase = half << 3;
    const int tm = blockIdx.x * 16;
    const int tn = blockIdx.y * 16;

    const int am  = tm + lm;
    const int bn  = tn + lm;
    const int amc = imin(am, M - 1);
    const int bnc = imin(bn, N - 1);
    const bool amv = (am < M), bnv = (bn < N);

    v8f acc = {};
    for (int k0 = 0; k0 < K; k0 += 32) {
        v16h a, b;
#pragma unroll
        for (int j = 0; j < 8; ++j) {
            const int kj = (j < 4) ? (kbase + 2 * j) : (16 + kbase + 2 * (j - 4));
#pragma unroll
            for (int t = 0; t < 2; ++t) {
                const int kk  = k0 + kj + t;
                const int kkc = imin(kk, K - 1);
                float av = A[(long long)amc * As_m + (long long)kkc * As_k];
                float bv = B[(long long)bnc * Bs_n + (long long)kkc * Bs_k];
                av = (amv && kk < K) ? av : 0.0f;
                bv = (bnv && kk < K) ? bv : 0.0f;
                a[2 * j + t] = (_Float16)av;
                b[2 * j + t] = (_Float16)bv;
            }
        }
        acc = __builtin_amdgcn_wmma_f32_16x16x32_f16(
                  false, a, false, b, (short)0, acc, false, false);
    }

    const int n = tn + lm;
    if (n < N) {
#pragma unroll
        for (int r = 0; r < 8; ++r) {
            const int m = tm + (half << 3) + r;
            if (m < M) {
                long long oi = (long long)m * Os_m + (long long)n * Os_n;
                float v = acc[r];
                if (Res) v += Res[oi];
                O[oi] = v;
            }
        }
    }
}

// ---------------------------------------------------------------------------
// Depthwise 3x3 conv (SAME) + bias + SiLU. xz row-major [L, 2*di]; u [di, L].
// ---------------------------------------------------------------------------
__global__ void dwconv2d_silu(const float* __restrict__ xz,
                              const float* __restrict__ w,
                              const float* __restrict__ bias,
                              float* __restrict__ u,
                              int H, int W, int di, int twodi)
{
    const int L = H * W;
    int idx = blockIdx.x * blockDim.x + threadIdx.x;
    if (idx >= di * L) return;
    int d = idx / L, l = idx - d * L;
    int h = l / W, wc = l - h * W;
    float acc = bias[d];
#pragma unroll
    for (int i = 0; i < 3; ++i)
#pragma unroll
        for (int j = 0; j < 3; ++j) {
            int hh = h + i - 1, ww = wc + j - 1;
            if (hh >= 0 && hh < H && ww >= 0 && ww < W)
                acc += w[d * 9 + i * 3 + j] *
                       xz[(long long)(hh * W + ww) * twodi + d];
        }
    u[(long long)d * L + l] = acc / (1.0f + expf(-acc));
}

__global__ void dwconv1d_silu(const float* __restrict__ xz,
                              const float* __restrict__ w,
                              const float* __restrict__ bias,
                              float* __restrict__ u,
                              int L, int di, int twodi)
{
    int idx = blockIdx.x * blockDim.x + threadIdx.x;
    if (idx >= di * L) return;
    int d = idx / L, l = idx - d * L;
    float acc = bias[d];
#pragma unroll
    for (int j = 0; j < 3; ++j) {
        int ll = l + j - 1;
        if (ll >= 0 && ll < L)
            acc += w[d * 3 + j] * xz[(long long)ll * twodi + d];
    }
    u[(long long)d * L + l] = acc / (1.0f + expf(-acc));
}

// ---------------------------------------------------------------------------
// Scan directions. 2-D: fwd row-major / fwd col-major / both reversed.
// ---------------------------------------------------------------------------
__global__ void build_xs_2d(const float* __restrict__ u, float* __restrict__ xs,
                            int di, int H, int W)
{
    const int L = H * W;
    int idx = blockIdx.x * blockDim.x + threadIdx.x;
    if (idx >= 4 * di * L) return;
    int k = idx / (di * L);
    int rem = idx - k * (di * L);
    int d = rem / L, l = rem - d * L;
    int src;
    if (k == 0)      src = l;
    else if (k == 1) src = (l % H) * W + (l / H);
    else if (k == 2) src = L - 1 - l;
    else { int lr = L - 1 - l; src = (lr % H) * W + (lr / H); }
    xs[idx] = u[(long long)d * L + src];
}

__global__ void build_xs_1d(const float* __restrict__ u, float* __restrict__ xs,
                            int di, int L)
{
    int idx = blockIdx.x * blockDim.x + threadIdx.x;
    if (idx >= 2 * di * L) return;
    int k = idx / (di * L);
    int rem = idx - k * (di * L);
    int d = rem / L, l = rem - d * L;
    int src = (k == 0) ? l : (L - 1 - l);
    xs[idx] = u[(long long)d * L + src];
}

// ---------------------------------------------------------------------------
// Selective scan. 16 lanes per (k,d) channel; lane n owns state h_n.
// ---------------------------------------------------------------------------
__global__ void scan_kernel(const float* __restrict__ xs,
                            const float* __restrict__ xdbl,
                            const float* __restrict__ dtraw,
                            const float* __restrict__ dt_b,
                            const float* __restrict__ A_log,
                            const float* __restrict__ Dp,
                            float* __restrict__ ys,
                            int K, int di, int L, int dr)
{
    const int C2 = dr + 2 * NST;
    int gid = blockIdx.x * blockDim.x + threadIdx.x;
    int n  = gid & 15;
    int ch = gid >> 4;
    if (ch >= K * di) return;
    int k = ch / di, d = ch - k * di;

    const float Acoef = -expf(A_log[((long long)k * di + d) * NST + n]);
    const float dbias = dt_b[k * di + d];
    const float Dv    = Dp[k * di + d];

    const float* xrow  = xs    + ((long long)k * di + d) * L;
    const float* dtrow = dtraw + ((long long)k * di + d) * L;
    const float* Bp    = xdbl  + ((long long)k * C2 + dr + n) * L;
    const float* Cp    = Bp + (long long)NST * L;
    float* yrow        = ys    + ((long long)k * di + d) * L;

    float h = 0.0f;
    for (int l = 0; l < L; ++l) {
        __builtin_prefetch(Bp + l + 128, 0, 1);   // global_prefetch_b8
        __builtin_prefetch(Cp + l + 128, 0, 1);
        float dp    = dtrow[l] + dbias;
        float delta = (dp > 20.0f) ? dp : log1pf(expf(dp));
        float x     = xrow[l];
        float dA    = expf(delta * Acoef);
        float bu    = delta * Bp[l] * x;
        h = dA * h + bu;
        float pc = h * Cp[l];
        pc += __shfl_xor(pc, 1, 16);
        pc += __shfl_xor(pc, 2, 16);
        pc += __shfl_xor(pc, 4, 16);
        pc += __shfl_xor(pc, 8, 16);
        if (n == 0) yrow[l] = pc + Dv * x;
    }
}

// ---------------------------------------------------------------------------
// Direction recombination.
// ---------------------------------------------------------------------------
__global__ void combine2d(const float* __restrict__ ys, float* __restrict__ yc,
                          int di, int H, int W)
{
    const int L = H * W;
    int idx = blockIdx.x * blockDim.x + threadIdx.x;
    if (idx >= L * di) return;
    int l = idx / di, d = idx - l * di;
    int h = l / W, w = l - h * W;
    long long s = (long long)di * L;
    long long base = (long long)d * L;
    int lp = w * H + h;
    float y0 = ys[0 * s + base + l];
    float y1 = ys[1 * s + base + lp];
    float y2 = ys[2 * s + base + (L - 1 - l)];
    float y3 = ys[3 * s + base + (L - 1 - lp)];
    yc[(long long)l * di + d] = y0 + y1 + y2 + y3;
}

__global__ void combine1d(const float* __restrict__ ys, float* __restrict__ yc,
                          int di, int L)
{
    int idx = blockIdx.x * blockDim.x + threadIdx.x;
    if (idx >= L * di) return;
    int l = idx / di, d = idx - l * di;
    long long s = (long long)di * L;
    long long base = (long long)d * L;
    yc[(long long)l * di + d] = ys[base + l] + ys[s + base + (L - 1 - l)];
}

// LayerNorm(row) * SiLU(z);  yc [L,di] row-major, z = xz[:, di:2di].
__global__ void ln_gate(const float* __restrict__ yc, const float* __restrict__ xz,
                        const float* __restrict__ g, const float* __restrict__ b,
                        float* __restrict__ out, int L, int di, int twodi)
{
    int l = blockIdx.x * blockDim.x + threadIdx.x;
    if (l >= L) return;
    const float* row = yc + (long long)l * di;
    float m = 0.0f;
    for (int e = 0; e < di; ++e) m += row[e];
    m /= di;
    float v = 0.0f;
    for (int e = 0; e < di; ++e) { float t = row[e] - m; v += t * t; }
    v /= di;
    float rs = rsqrtf(v + 1e-5f);
    for (int e = 0; e < di; ++e) {
        float z = xz[(long long)l * twodi + di + e];
        float gate = z / (1.0f + expf(-z));
        out[(long long)l * di + e] = ((row[e] - m) * rs * g[e] + b[e]) * gate;
    }
}

__global__ void ln_rows(const float* __restrict__ X, const float* __restrict__ g,
                        const float* __restrict__ b, float* __restrict__ O,
                        int L, int C)
{
    int l = blockIdx.x * blockDim.x + threadIdx.x;
    if (l >= L) return;
    const float* row = X + (long long)l * C;
    float m = 0.0f;
    for (int e = 0; e < C; ++e) m += row[e];
    m /= C;
    float v = 0.0f;
    for (int e = 0; e < C; ++e) { float t = row[e] - m; v += t * t; }
    v /= C;
    float rs = rsqrtf(v + 1e-5f);
    for (int e = 0; e < C; ++e)
        O[(long long)l * C + e] = (row[e] - m) * rs * g[e] + b[e];
}

// ---------------------------------------------------------------------------
// Host-side orchestration
// ---------------------------------------------------------------------------
struct ModP {
    const float *in_w, *conv_w, *conv_b, *xproj_w, *dt_w, *dt_b,
                *A_log, *D, *ln_g, *ln_b, *out_w;
};

static inline ModP getP(void* const* d_in, int base)
{
    ModP p;
    p.in_w    = (const float*)d_in[base + 0];
    p.conv_w  = (const float*)d_in[base + 1];
    p.conv_b  = (const float*)d_in[base + 2];
    p.xproj_w = (const float*)d_in[base + 3];
    p.dt_w    = (const float*)d_in[base + 4];
    p.dt_b    = (const float*)d_in[base + 5];
    p.A_log   = (const float*)d_in[base + 6];
    p.D       = (const float*)d_in[base + 7];
    p.ln_g    = (const float*)d_in[base + 8];
    p.ln_b    = (const float*)d_in[base + 9];
    p.out_w   = (const float*)d_in[base + 10];
    return p;
}

// Every GEMM operand here has a unit stride on exactly one axis, so the
// TDM-staged kernel applies to all of them.
static inline void gemm(hipStream_t s, const float* A, const float* B, float* O,
                        const float* Res, int M, int N, int K,
                        long long As_m, long long As_k,
                        long long Bs_n, long long Bs_k,
                        long long Os_m, long long Os_n)
{
    int aRM = (As_k == 1);
    int bRM = (Bs_k == 1);
    long long Abig = aRM ? As_m : As_k;
    long long Bbig = bRM ? Bs_n : Bs_k;
    dim3 g((M + 15) / 16, (N + 15) / 16), b(32);
    wmma_gemm_tdm<<<g, b, 0, s>>>(A, B, O, Res, M, N, K,
                                  Abig, aRM, Bbig, bRM, Os_m, Os_n);
}

static inline int cdiv(int a, int b) { return (a + b - 1) / b; }

static void run_ss2d_2d(hipStream_t s, const float* X, int dm, int H, int W,
                        const ModP& p, float* Fout, float* arena)
{
    const int di = 2 * dm, L = H * W, K = 4;
    const int dr = (dm + 15) / 16, C2 = dr + 2 * NST;

    float* xz    = arena;                 size_t o = (size_t)L * 2 * di;
    float* u     = arena + o;             o += (size_t)di * L;
    float* xs    = arena + o;             o += (size_t)K * di * L;
    float* xdbl  = arena + o;             o += (size_t)K * C2 * L;
    float* dtraw = arena + o;             o += (size_t)K * di * L;
    float* ys    = arena + o;             o += (size_t)K * di * L;
    float* yc    = arena + o;
    float* gated = u;                     // u dead after xs is built

    // in-proj: xz[l,e] = sum_c X[c,l] * in_w[e,c]
    gemm(s, X, p.in_w, xz, nullptr, L, 2 * di, dm,
         1LL, (long long)L, (long long)dm, 1LL, (long long)(2 * di), 1LL);

    int t = di * L;
    dwconv2d_silu<<<cdiv(t, 256), 256, 0, s>>>(xz, p.conv_w, p.conv_b, u, H, W, di, 2 * di);

    t = 4 * di * L;
    build_xs_2d<<<cdiv(t, 256), 256, 0, s>>>(u, xs, di, H, W);

    for (int k = 0; k < K; ++k) {
        gemm(s, p.xproj_w + (size_t)k * C2 * di, xs + (size_t)k * di * L,
             xdbl + (size_t)k * C2 * L, nullptr, C2, L, di,
             (long long)di, 1LL, 1LL, (long long)L, (long long)L, 1LL);
        gemm(s, p.dt_w + (size_t)k * di * dr, xdbl + (size_t)k * C2 * L,
             dtraw + (size_t)k * di * L, nullptr, di, L, dr,
             (long long)dr, 1LL, 1LL, (long long)L, (long long)L, 1LL);
    }

    t = K * di * NST;
    scan_kernel<<<cdiv(t, 256), 256, 0, s>>>(xs, xdbl, dtraw, p.dt_b, p.A_log,
                                             p.D, ys, K, di, L, dr);

    t = L * di;
    combine2d<<<cdiv(t, 256), 256, 0, s>>>(ys, yc, di, H, W);

    ln_gate<<<cdiv(L, 256), 256, 0, s>>>(yc, xz, p.ln_g, p.ln_b, gated, L, di, 2 * di);

    // out-proj -> NCHW flat: Fout[c*L + l]
    gemm(s, gated, p.out_w, Fout, nullptr, L, dm, di,
         (long long)di, 1LL, (long long)di, 1LL, 1LL, (long long)L);
}

static void run_ss2d_1d(hipStream_t s, const float* Fln, const float* Fres,
                        int dm, int L, const ModP& p, float* Out, float* arena)
{
    const int di = 2 * dm, K = 2;
    const int dr = (dm + 15) / 16, C2 = dr + 2 * NST;

    float* xz    = arena;                 size_t o = (size_t)L * 2 * di;
    float* u     = arena + o;             o += (size_t)di * L;
    float* xs    = arena + o;             o += (size_t)K * di * L;
    float* xdbl  = arena + o;             o += (size_t)K * C2 * L;
    float* dtraw = arena + o;             o += (size_t)K * di * L;
    float* ys    = arena + o;             o += (size_t)K * di * L;
    float* yc    = arena + o;
    float* gated = u;

    gemm(s, Fln, p.in_w, xz, nullptr, L, 2 * di, dm,
         (long long)dm, 1LL, (long long)dm, 1LL, (long long)(2 * di), 1LL);

    int t = di * L;
    dwconv1d_silu<<<cdiv(t, 256), 256, 0, s>>>(xz, p.conv_w, p.conv_b, u, L, di, 2 * di);

    t = 2 * di * L;
    build_xs_1d<<<cdiv(t, 256), 256, 0, s>>>(u, xs, di, L);

    for (int k = 0; k < K; ++k) {
        gemm(s, p.xproj_w + (size_t)k * C2 * di, xs + (size_t)k * di * L,
             xdbl + (size_t)k * C2 * L, nullptr, C2, L, di,
             (long long)di, 1LL, 1LL, (long long)L, (long long)L, 1LL);
        gemm(s, p.dt_w + (size_t)k * di * dr, xdbl + (size_t)k * C2 * L,
             dtraw + (size_t)k * di * L, nullptr, di, L, dr,
             (long long)dr, 1LL, 1LL, (long long)L, (long long)L, 1LL);
    }

    t = K * di * NST;
    scan_kernel<<<cdiv(t, 256), 256, 0, s>>>(xs, xdbl, dtraw, p.dt_b, p.A_log,
                                             p.D, ys, K, di, L, dr);

    t = L * di;
    combine1d<<<cdiv(t, 256), 256, 0, s>>>(ys, yc, di, L);

    ln_gate<<<cdiv(L, 256), 256, 0, s>>>(yc, xz, p.ln_g, p.ln_b, gated, L, di, 2 * di);

    // out-proj with residual
    gemm(s, gated, p.out_w, Out, Fres, L, dm, di,
         (long long)di, 1LL, (long long)di, 1LL, (long long)dm, 1LL);
}

extern "C" void kernel_launch(void* const* d_in, const int* in_sizes, int n_in,
                              void* d_out, int out_size, void* d_ws, size_t ws_size,
                              hipStream_t stream)
{
    (void)in_sizes; (void)n_in; (void)out_size; (void)ws_size;

    const float* C1 = (const float*)d_in[0];   // [24,112,112]
    const float* C2 = (const float*)d_in[1];   // [48,56,56]
    const float* C3 = (const float*)d_in[2];   // [96,28,28]
    ModP p1 = getP(d_in, 3);
    ModP p2 = getP(d_in, 14);
    ModP p3 = getP(d_in, 25);
    ModP pf = getP(d_in, 36);
    const float* n1_g = (const float*)d_in[47];
    const float* n1_b = (const float*)d_in[48];

    const int Lf = 21952;            // 12544 + 6272 + 3136 rows of 24
    const size_t Fsz = (size_t)Lf * 24;

    float* ws    = (float*)d_ws;
    float* F     = ws;               // fused sequence, flat NCHW concat
    float* Fln   = ws + Fsz;         // LN(F)
    float* arena = ws + 2 * Fsz;     // per-module scratch (reused serially)

    run_ss2d_2d(stream, C1, 24, 112, 112, p1, F,          arena);
    run_ss2d_2d(stream, C2, 48,  56,  56, p2, F + 301056, arena);
    run_ss2d_2d(stream, C3, 96,  28,  28, p3, F + 451584, arena);

    ln_rows<<<cdiv(Lf, 256), 256, 0, stream>>>(F, n1_g, n1_b, Fln, Lf, 24);

    run_ss2d_1d(stream, Fln, F, 24, Lf, pf, (float*)d_out, arena);
}
